// GlobalSAModule_12060268167709
// MI455X (gfx1250) — matrix operand order repaired and checked by reading, not compiled, not run on hardware
//
#include <hip/hip_runtime.h>

typedef __attribute__((ext_vector_type(16))) __bf16 v16bf;
typedef __attribute__((ext_vector_type(8)))  __bf16 v8bf;
typedef __attribute__((ext_vector_type(4)))  __bf16 v4bf;
typedef __attribute__((ext_vector_type(8)))  float  v8f;
typedef __attribute__((ext_vector_type(4)))  unsigned int v4u;
typedef __attribute__((ext_vector_type(8)))  int    v8i;
typedef __attribute__((ext_vector_type(4)))  int    v4i;

#if defined(__has_builtin)
#if __has_builtin(__builtin_amdgcn_tensor_load_to_lds)
#define HAVE_TDM 1
#endif
#endif
#ifndef HAVE_TDM
#define HAVE_TDM 0
#endif

// ---------- order-preserving float<->uint mapping (for atomic max) ----------
__device__ __forceinline__ unsigned mapOrd(float f) {
  unsigned u = __float_as_uint(f);
  return (u & 0x80000000u) ? ~u : (u | 0x80000000u);
}
__device__ __forceinline__ float unmapOrd(unsigned u) {
  unsigned b = (u & 0x80000000u) ? (u ^ 0x80000000u) : ~u;
  return __uint_as_float(b);
}

#if HAVE_TDM
// 2D tile load via Tensor Data Mover. Units are 8-byte elements (data_size=3).
// D# layout per CDNA5 ISA 8.3/8.4 (group0: count/lds/global/type, group1: dims).
__device__ __forceinline__ void tdmLoad2D(const void* gsrc, unsigned ldsOff,
                                          unsigned tensorDim0, unsigned rows,
                                          unsigned tileDim0, unsigned stride) {
  unsigned long long ga = reinterpret_cast<unsigned long long>(gsrc);
  v4u g0;
  g0[0] = 1u;                                    // count=1, user-mode descriptor
  g0[1] = ldsOff;                                // lds_addr (bytes)
  g0[2] = (unsigned)ga;                          // global_addr[31:0]
  g0[3] = ((unsigned)(ga >> 32) & 0x01FFFFFFu)   // global_addr[56:32]
          | (2u << 30);                          // type = 2 ("image")
  v8i g1;
  g1[0] = (int)(3u << 16);                               // data_size = 3 (8B)
  g1[1] = (int)(tensorDim0 << 16);                       // tensor_dim0 lo @[63:48]
  g1[2] = (int)((tensorDim0 >> 16) | (rows << 16));      // dim0 hi | tensor_dim1 lo
  g1[3] = (int)((rows >> 16) | (tileDim0 << 16));        // dim1 hi | tile_dim0
  g1[4] = (int)(rows & 0xFFFFu);                         // tile_dim1 | tile_dim2=0
  g1[5] = (int)stride;                                   // tensor_dim0_stride lo32
  g1[6] = 0;
  g1[7] = 0;
  v4i g2; g2[0] = 0; g2[1] = 0; g2[2] = 0; g2[3] = 0;
  v4i g3; g3[0] = 0; g3[1] = 0; g3[2] = 0; g3[3] = 0;
#if __clang_major__ >= 23
  v8i g4; g4[0]=0; g4[1]=0; g4[2]=0; g4[3]=0; g4[4]=0; g4[5]=0; g4[6]=0; g4[7]=0;
  __builtin_amdgcn_tensor_load_to_lds(g0, g1, g2, g3, g4, 0);
#else
  __builtin_amdgcn_tensor_load_to_lds(g0, g1, g2, g3, 0);
#endif
}
#endif

// ---------------------------- init ----------------------------
__global__ void k_init(unsigned* __restrict__ outU, float* __restrict__ posSum,
                       float* __restrict__ counts,
                       unsigned long long* __restrict__ keys, int B) {
  int i = blockIdx.x * blockDim.x + threadIdx.x;
  if (i < B * 128) outU[i] = 0x00800000u;  // mapOrd(-FLT_MAX)
  if (i < 3 * B)   posSum[i] = 0.f;
  if (i < B)       { counts[i] = 0.f; keys[i] = ~0ull; }
}

// ------------------- center-of-mass accumulation -------------------
__global__ void k_com_accum(const float* __restrict__ pos, const int* __restrict__ batch,
                            float* __restrict__ posSum, float* __restrict__ counts,
                            int N, int B) {
  __shared__ float ls[256];  // 4 slots per graph, B <= 64
  for (int j = threadIdx.x; j < 4 * B; j += blockDim.x) ls[j] = 0.f;
  __syncthreads();
  int i = blockIdx.x * blockDim.x + threadIdx.x;
  if (i < N) {
    int g = batch[i];
    atomicAdd(&ls[g * 4 + 0], pos[i * 3 + 0]);
    atomicAdd(&ls[g * 4 + 1], pos[i * 3 + 1]);
    atomicAdd(&ls[g * 4 + 2], pos[i * 3 + 2]);
    atomicAdd(&ls[g * 4 + 3], 1.f);
  }
  __syncthreads();
  for (int j = threadIdx.x; j < 4 * B; j += blockDim.x) {
    float v = ls[j];
    if (v != 0.f) {
      int g = j >> 2, d = j & 3;
      if (d < 3) atomicAdd(&posSum[g * 3 + d], v);
      else       atomicAdd(&counts[g], v);
    }
  }
}

__global__ void k_com_final(float* __restrict__ coms, const float* __restrict__ posSum,
                            const float* __restrict__ counts, int B) {
  int i = blockIdx.x * blockDim.x + threadIdx.x;
  if (i < 3 * B) coms[i] = posSum[i] / fmaxf(counts[i / 3], 1.f);
}

// ------------------- scatter-argmin (first occurrence) -------------------
__global__ void k_argmin(const float* __restrict__ pos, const int* __restrict__ batch,
                         const float* __restrict__ coms,
                         unsigned long long* __restrict__ keys, int N, int B) {
  __shared__ unsigned long long lk[64];
  for (int j = threadIdx.x; j < B; j += blockDim.x) lk[j] = ~0ull;
  __syncthreads();
  int i = blockIdx.x * blockDim.x + threadIdx.x;
  if (i < N) {
    int g = batch[i];
    float dx = pos[i * 3 + 0] - coms[g * 3 + 0];
    float dy = pos[i * 3 + 1] - coms[g * 3 + 1];
    float dz = pos[i * 3 + 2] - coms[g * 3 + 2];
    float d2 = dx * dx + dy * dy + dz * dz;  // sqrt is monotone: same argmin/ties
    unsigned long long key =
        ((unsigned long long)__float_as_uint(d2) << 32) | (unsigned)i;
    atomicMin(&lk[g], key);
  }
  __syncthreads();
  for (int j = threadIdx.x; j < B; j += blockDim.x)
    if (lk[j] != ~0ull) atomicMin(&keys[j], lk[j]);
}

// ------------------- gather destination-node data -------------------
__global__ void k_gather(const float* __restrict__ x, const float* __restrict__ pos,
                         const int* __restrict__ batch, const float* __restrict__ lfr,
                         const unsigned long long* __restrict__ keys,
                         float* __restrict__ xd, float* __restrict__ posDst,
                         float* __restrict__ outF, int N, int B) {
  int b = blockIdx.x, t = threadIdx.x;
  unsigned id32 = (unsigned)(keys[b] & 0xffffffffull);
  int idx = (id32 >= (unsigned)N) ? (N - 1) : (int)id32;
  if (t < 64) xd[b * 64 + t] = x[idx * 64 + t];
  if (t < 3) {
    float p = pos[idx * 3 + t];
    posDst[b * 3 + t] = p;
    outF[B * 128 + b * 3 + t] = p;           // pos_dst output
  }
  if (t == 0) outF[B * 128 + 3 * B + b] = (float)batch[idx];  // batch[idx] output
  if (t < 9)  outF[B * 128 + 4 * B + b * 9 + t] = lfr[idx * 9 + t];  // lframes_dst
}

// --------- weights -> bf16, TRANSPOSED [n][k] (W1 K-padded 131 -> 160) ---------
__global__ void k_wconv(const float* __restrict__ W1, const float* __restrict__ W2,
                        __bf16* __restrict__ W1bT, __bf16* __restrict__ W2bT) {
  int i = blockIdx.x * blockDim.x + threadIdx.x;
  if (i < 128 * 160) {
    int n = i / 160, k = i % 160;
    W1bT[i] = (__bf16)((k < 131) ? W1[k * 128 + n] : 0.f);
  } else if (i < 128 * 160 + 128 * 128) {
    int j = i - 128 * 160;
    int n = j >> 7, k = j & 127;
    W2bT[j] = (__bf16)W2[k * 128 + n];
  }
}

// ------------------- fused MLP (2x WMMA GEMM) + segment-max -------------------
// Dynamic LDS layout (base offset 0: no static LDS in this kernel):
//   [0      , 32768) h_s   bf16[128][128]  hidden activations
//   [32768  , 40960) wT0   bf16[128][32]   weight chunk buffer 0 ([n][k])
//   [40960  , 49152) wT1   bf16[128][32]   weight chunk buffer 1
//   [49152  , 57344) a_s   bf16[128][32]   A (feat) chunk ([row][k])
//   [57344  , 57856) rb    int[128]
//   [57856  , 59904) omax  uint[4][128]
#define SMEM_BYTES 59904
#define LDS_WT0 32768u
#define LDS_WT1 40960u

__global__ __launch_bounds__(256) void k_main(
    const float* __restrict__ x, const float* __restrict__ pos,
    const int* __restrict__ batch, const float* __restrict__ b1,
    const float* __restrict__ b2, const float* __restrict__ xd,
    const float* __restrict__ posDst, const __bf16* __restrict__ W1bT,
    const __bf16* __restrict__ W2bT, unsigned* __restrict__ outU, int N) {
  extern __shared__ char smem[];
  __bf16* h_s = (__bf16*)smem;
  __bf16* wT0 = (__bf16*)(smem + LDS_WT0);
  __bf16* a_s = (__bf16*)(smem + 49152);
  int* rb = (int*)(smem + 57344);
  unsigned* omax = (unsigned*)(smem + 57856);

  const int tid  = threadIdx.x;
  const int lane = tid & 31;
  const int wv   = tid >> 5;       // wave id == M-tile id (0..7)
  const int half = lane >> 4;      // 0: lanes 0-15, 1: lanes 16-31
  const int Mrow = lane & 15;
  const int Ncol = lane & 15;
  const int kb   = half * 16;      // A-frag K base (contiguous, matches B)
  const int ko   = half * 16;      // B-frag K base
  const int r0   = blockIdx.x * 128;

  for (int j = tid; j < 4 * 128; j += 256) omax[j] = 0u;
  if (tid < 128) {
    int r = r0 + tid;
    rb[tid] = (r < N) ? batch[r] : -1;
  }
  __syncthreads();

#if HAVE_TDM
  // preload W1 chunk 0 (tile: 128 rows x 64B, row stride 320B, 8B units)
  if (wv == 0) tdmLoad2D(W1bT, LDS_WT0, 40u, 128u, 8u, 40u);
#endif

  v8f acc[8];
  #pragma unroll
  for (int nt = 0; nt < 8; ++nt)
    #pragma unroll
    for (int e = 0; e < 8; ++e) acc[nt][e] = 0.f;

  // ---------------- GEMM1: feat[128x160] @ W1[160x128] ----------------
  for (int kc = 0; kc < 5; ++kc) {
#if HAVE_TDM
    const int buf = kc & 1;
#else
    const int buf = 0;
    __syncthreads();
    for (int u = tid; u < 128 * 4; u += 256) {   // vectorized chunk copy
      int n = u >> 2, s = u & 3;
      *(v8bf*)&wT0[n * 32 + s * 8] =
          *(const v8bf*)&W1bT[n * 160 + kc * 32 + s * 8];
    }
#endif
    // build feat chunk on the fly: [xd | x - xd | pos - pos_dst | 0-pad]
    const int cbase = kc * 32;
    if (kc < 4) {  // dense 4-wide path (columns 0..127)
      for (int u = tid; u < 128 * 8; u += 256) {
        int row = u >> 3, q = u & 7;
        int c = cbase + q * 4;
        int g = rb[row];
        float v0 = 0.f, v1 = 0.f, v2 = 0.f, v3 = 0.f;
        if (g >= 0) {
          if (c < 64) {
            float4 dv = *(const float4*)&xd[g * 64 + c];
            v0 = dv.x; v1 = dv.y; v2 = dv.z; v3 = dv.w;
          } else {
            int r = r0 + row;
            float4 xv = *(const float4*)&x[r * 64 + (c - 64)];
            float4 dv = *(const float4*)&xd[g * 64 + (c - 64)];
            v0 = xv.x - dv.x; v1 = xv.y - dv.y;
            v2 = xv.z - dv.z; v3 = xv.w - dv.w;
          }
        }
        v4bf o; o[0] = (__bf16)v0; o[1] = (__bf16)v1;
        o[2] = (__bf16)v2; o[3] = (__bf16)v3;
        *(v4bf*)&a_s[row * 32 + q * 4] = o;
      }
    } else {       // tail chunk: columns 128..159 (only 128..130 nonzero)
      for (int j = tid; j < 128 * 32; j += 256) {
        int row = j >> 5, c0 = j & 31, c = cbase + c0;
        int g = rb[row];
        float v = 0.f;
        if (g >= 0 && c < 131) {
          int r = r0 + row;
          v = pos[r * 3 + (c - 128)] - posDst[g * 3 + (c - 128)];
        }
        a_s[row * 32 + c0] = (__bf16)v;
      }
    }
#if HAVE_TDM
    if (wv == 0) {
      if (kc + 1 < 5) {  // keep one DMA in flight; wait for current chunk
        tdmLoad2D((const char*)W1bT + (kc + 1) * 64,
                  (kc & 1) ? LDS_WT0 : LDS_WT1, 40u, 128u, 8u, 40u);
        __builtin_amdgcn_s_wait_tensorcnt(1);
      } else {
        __builtin_amdgcn_s_wait_tensorcnt(0);
      }
    }
#endif
    __syncthreads();
    const __bf16* ar = &a_s[(wv * 16 + Mrow) * 32];
    v8bf alo = *reinterpret_cast<const v8bf*>(ar + kb);
    v8bf ahi = *reinterpret_cast<const v8bf*>(ar + 8 + kb);
    v16bf afrag;
    #pragma unroll
    for (int e = 0; e < 8; ++e) { afrag[e] = alo[e]; afrag[8 + e] = ahi[e]; }
    const __bf16* wp = (const __bf16*)(smem + LDS_WT0 + (unsigned)buf * 8192u);
    v16bf bfrag[8];
    #pragma unroll
    for (int nt = 0; nt < 8; ++nt)   // batch all B-frag loads: one dscnt drain
      bfrag[nt] = *reinterpret_cast<const v16bf*>(&wp[(nt * 16 + Ncol) * 32 + ko]);
    #pragma unroll
    for (int nt = 0; nt < 8; ++nt)   // then issue WMMAs back-to-back
      acc[nt] = __builtin_amdgcn_wmma_f32_16x16x32_bf16(
          false, afrag, false, bfrag[nt], (short)0, acc[nt], false, false);
    __syncthreads();
  }

#if HAVE_TDM
  // preload W2 chunk 0 while bias/ReLU runs (wT0 free: barrier above)
  if (wv == 0) tdmLoad2D(W2bT, LDS_WT0, 32u, 128u, 8u, 32u);
#endif
  // bias + ReLU -> h_s (bf16); reset accumulators for GEMM2
  #pragma unroll
  for (int nt = 0; nt < 8; ++nt) {
    int col = nt * 16 + Ncol;
    float bb = b1[col];
    #pragma unroll
    for (int v = 0; v < 8; ++v) {
      float hv = fmaxf(acc[nt][v] + bb, 0.f);
      h_s[(wv * 16 + v + 8 * half) * 128 + col] = (__bf16)hv;
      acc[nt][v] = 0.f;
    }
  }
  __syncthreads();

  // ---------------- GEMM2: h[128x128] @ W2[128x128] ----------------
  for (int kc = 0; kc < 4; ++kc) {
#if HAVE_TDM
    const int buf = kc & 1;
    if (wv == 0) {
      if (kc + 1 < 4) {
        tdmLoad2D((const char*)W2bT + (kc + 1) * 64,
                  (kc & 1) ? LDS_WT0 : LDS_WT1, 32u, 128u, 8u, 32u);
        __builtin_amdgcn_s_wait_tensorcnt(1);
      } else {
        __builtin_amdgcn_s_wait_tensorcnt(0);
      }
    }
#else
    const int buf = 0;
    for (int u = tid; u < 128 * 4; u += 256) {
      int n = u >> 2, s = u & 3;
      *(v8bf*)&wT0[n * 32 + s * 8] =
          *(const v8bf*)&W2bT[n * 128 + kc * 32 + s * 8];
    }
#endif
    __syncthreads();
    const __bf16* ar = &h_s[(wv * 16 + Mrow) * 128 + kc * 32];
    v8bf alo = *reinterpret_cast<const v8bf*>(ar + kb);
    v8bf ahi = *reinterpret_cast<const v8bf*>(ar + 8 + kb);
    v16bf afrag;
    #pragma unroll
    for (int e = 0; e < 8; ++e) { afrag[e] = alo[e]; afrag[8 + e] = ahi[e]; }
    const __bf16* wp = (const __bf16*)(smem + LDS_WT0 + (unsigned)buf * 8192u);
    v16bf bfrag[8];
    #pragma unroll
    for (int nt = 0; nt < 8; ++nt)
      bfrag[nt] = *reinterpret_cast<const v16bf*>(&wp[(nt * 16 + Ncol) * 32 + ko]);
    #pragma unroll
    for (int nt = 0; nt < 8; ++nt)
      acc[nt] = __builtin_amdgcn_wmma_f32_16x16x32_bf16(
          false, afrag, false, bfrag[nt], (short)0, acc[nt], false, false);
    __syncthreads();
  }

  // ---------------- segmented max (batch sorted within block) ----------------
  const int gBlk = rb[0];
  const int gTa = rb[wv * 16];
  const int gTb = rb[wv * 16 + 15];
  const bool uni = (gTa == gTb) && (gTa >= 0);  // wave-uniform
  #pragma unroll
  for (int nt = 0; nt < 8; ++nt) {
    int col = nt * 16 + Ncol;
    float b2v = b2[col];
    if (uni) {
      float m = acc[nt][0];
      #pragma unroll
      for (int v = 1; v < 8; ++v) m = fmaxf(m, acc[nt][v]);
      m = fmaxf(m, __shfl_xor(m, 16, 32));  // combine the two row-halves
      if (half == 0) {
        unsigned u = mapOrd(m + b2v);
        int slot = gTa - gBlk;
        if (slot < 4) atomicMax(&omax[slot * 128 + col], u);
        else          atomicMax(&outU[gTa * 128 + col], u);
      }
    } else {
      #pragma unroll
      for (int v = 0; v < 8; ++v) {
        int g = rb[wv * 16 + v + 8 * half];
        if (g >= 0) {
          unsigned u = mapOrd(acc[nt][v] + b2v);
          int slot = g - gBlk;
          if (slot < 4) atomicMax(&omax[slot * 128 + col], u);
          else          atomicMax(&outU[g * 128 + col], u);
        }
      }
    }
  }
  __syncthreads();
  int nv = N - r0; if (nv > 128) nv = 128;
  const int gLast = rb[nv - 1];
  int nslots = gLast - gBlk + 1; if (nslots > 4) nslots = 4;
  for (int s = tid >> 7; s < nslots; s += 2) {
    int col = tid & 127;
    unsigned u = omax[s * 128 + col];
    if (u) atomicMax(&outU[(gBlk + s) * 128 + col], u);
  }
}

// ------------------- unmap ordered uints -> floats (in place) -------------------
__global__ void k_unmap(unsigned* __restrict__ outU, int B) {
  int i = blockIdx.x * blockDim.x + threadIdx.x;
  if (i < B * 128) {
    float f = unmapOrd(outU[i]);
    reinterpret_cast<float*>(outU)[i] = f;
  }
}

extern "C" void kernel_launch(void* const* d_in, const int* in_sizes, int n_in,
                              void* d_out, int out_size, void* d_ws, size_t ws_size,
                              hipStream_t stream) {
  const float* x   = (const float*)d_in[0];
  const float* pos = (const float*)d_in[1];
  const int*   bt  = (const int*)d_in[2];
  const float* lfr = (const float*)d_in[3];
  const float* W1  = (const float*)d_in[4];
  const float* b1  = (const float*)d_in[5];
  const float* W2  = (const float*)d_in[6];
  const float* b2  = (const float*)d_in[7];
  const int N = in_sizes[2];
  const int B = out_size / 141;  // 128 + 3 + 1 + 9 per graph

  char* ws = (char*)d_ws;
  size_t o = 0;
  float* posSum = (float*)(ws + o); o += (size_t)3 * B * 4;
  float* counts = (float*)(ws + o); o += (size_t)B * 4;
  float* coms   = (float*)(ws + o); o += (size_t)3 * B * 4;
  o = (o + 15) & ~(size_t)15;
  unsigned long long* keys = (unsigned long long*)(ws + o); o += (size_t)B * 8;
  float* xd     = (float*)(ws + o); o += (size_t)B * 64 * 4;
  float* posDst = (float*)(ws + o); o += (size_t)3 * B * 4;
  o = (o + 15) & ~(size_t)15;
  __bf16* W1bT  = (__bf16*)(ws + o); o += (size_t)128 * 160 * 2;
  __bf16* W2bT  = (__bf16*)(ws + o); o += (size_t)128 * 128 * 2;

  float*    outF = (float*)d_out;
  unsigned* outU = (unsigned*)d_out;

  int nb256 = (N + 255) / 256;
  k_init<<<(B * 128 + 255) / 256, 256, 0, stream>>>(outU, posSum, counts, keys, B);
  k_com_accum<<<nb256, 256, 0, stream>>>(pos, bt, posSum, counts, N, B);
  k_com_final<<<(3 * B + 127) / 128, 128, 0, stream>>>(coms, posSum, counts, B);
  k_argmin<<<nb256, 256, 0, stream>>>(pos, bt, coms, keys, N, B);
  k_gather<<<B, 64, 0, stream>>>(x, pos, bt, lfr, keys, xd, posDst, outF, N, B);
  k_wconv<<<(128 * 160 + 128 * 128 + 255) / 256, 256, 0, stream>>>(W1, W2, W1bT, W2bT);
  k_main<<<(N + 127) / 128, 256, SMEM_BYTES, stream>>>(x, pos, bt, b1, b2, xd,
                                                       posDst, W1bT, W2bT, outU, N);
  k_unmap<<<(B * 128 + 255) / 256, 256, 0, stream>>>(outU, B);
}
